// DistributedExpert_48069273977257
// MI455X (gfx1250) — matrix operand order repaired and checked by reading, not compiled, not run on hardware
//
#include <hip/hip_runtime.h>
#include <hip/hip_bf16.h>
#include <math.h>

// ---------------------------------------------------------------------------
// Fixed problem sizes from the reference
// ---------------------------------------------------------------------------
#define SEQ_    4096
#define HID_    2048
#define EXP_    8192

typedef __attribute__((ext_vector_type(16))) __bf16 v16bf;
typedef __attribute__((ext_vector_type(8)))  __bf16 bf16x8;
typedef __attribute__((ext_vector_type(8)))  float  v8f;

// ---------------------------------------------------------------------------
// Fragment-major tiled layouts (bf16), 1KB per 16x32(A) / 32x16(B) tile:
//
// ALay(M,K): tile (tm,tk) at ((tm*(K/32)+tk)*512 elems). Element (m,kk):
//   elem = tile*512 + ((kk/8)*16 + m)*8 + kk%8
//   => wave A-frag load: lane l reads 16B at tile+(l+(l&16))*16B and 256B further on
//      (lane half h holds k = 16h+s, same order as B-frag) - coalesced.
//
// BLay(N,K): tile (tn,tk) at ((tn*(K/32)+tk)*512 elems). Element (n,kk):
//   elem = tile*512 + ((kk/16)*16 + n)*16 + kk%16
//   => wave B-frag load: lane l reads 32B at tile + l*32B - coalesced.
// ---------------------------------------------------------------------------

// f32 row-major [M,K] -> ALay bf16. One 16B chunk (8 elems) per thread.
__global__ void swizzleA_f32(const float* __restrict__ src,
                             __bf16* __restrict__ dst, int K, long long nChunks) {
    long long g = (long long)blockIdx.x * blockDim.x + threadIdx.x;
    if (g >= nChunks) return;
    const int numKT = K >> 5;
    long long T = g >> 6; int c = (int)(g & 63);
    int tm = (int)(T / numKT), tk = (int)(T % numKT);
    int kq = c >> 4, m = c & 15;
    const float* s = src + (size_t)(tm * 16 + m) * K + tk * 32 + kq * 8;
    float4 f0 = *(const float4*)(s);
    float4 f1 = *(const float4*)(s + 4);
    bf16x8 o;
    o[0]=(__bf16)f0.x; o[1]=(__bf16)f0.y; o[2]=(__bf16)f0.z; o[3]=(__bf16)f0.w;
    o[4]=(__bf16)f1.x; o[5]=(__bf16)f1.y; o[6]=(__bf16)f1.z; o[7]=(__bf16)f1.w;
    *(bf16x8*)(dst + g * 8) = o;
}

// f32 row-major [N,K] -> BLay bf16. One 32B chunk (16 elems) per thread.
__global__ void swizzleB_f32(const float* __restrict__ src,
                             __bf16* __restrict__ dst, int K, long long nChunks) {
    long long g = (long long)blockIdx.x * blockDim.x + threadIdx.x;
    if (g >= nChunks) return;
    const int numKT = K >> 5;
    long long T = g >> 5; int c = (int)(g & 31);
    int tn = (int)(T / numKT), tk = (int)(T % numKT);
    int kh = c >> 4, n = c & 15;
    const float* s = src + (size_t)(tn * 16 + n) * K + tk * 32 + kh * 16;
    __bf16* d = dst + g * 16;
#pragma unroll
    for (int u = 0; u < 2; ++u) {
        float4 f0 = *(const float4*)(s + u * 8);
        float4 f1 = *(const float4*)(s + u * 8 + 4);
        bf16x8 o;
        o[0]=(__bf16)f0.x; o[1]=(__bf16)f0.y; o[2]=(__bf16)f0.z; o[3]=(__bf16)f0.w;
        o[4]=(__bf16)f1.x; o[5]=(__bf16)f1.y; o[6]=(__bf16)f1.z; o[7]=(__bf16)f1.w;
        *(bf16x8*)(d + u * 8) = o;
    }
}

__device__ __forceinline__ float gelu_exact(float x) {
    return 0.5f * x * (1.0f + erff(x * 0.70710678118654752f));
}

// ---------------------------------------------------------------------------
// WMMA GEMM:  C[M,N] = act( A[M,K] * B[N,K]^T + bias ) (+res)
// A in ALay(M,K), B in BLay(N,K).
// Block: 256 thr = 8 wave32 (2 along M x 4 along N). Block tile 128x256,
// wave tile 64x64 = 4x4 v_wmma_f32_16x16x32_bf16 tiles. All operand loads
// are coalesced 512B/1KB tile streams.
// OMODE: 0 = f32 row-major
//        1 = bf16 ALay of [M,N] (k-dim = N)
//        2 = bf16 BLay of [M,N] (rows are n, k-dim = N)
//        3 = bf16 BLay of transpose: tensor [N,M], n = col, k-dim = M
// ---------------------------------------------------------------------------
template <int OMODE, bool DO_GELU, bool HAS_BIAS, bool HAS_RES>
__global__ __launch_bounds__(256) void gemm_bf16_wmma(
    const __bf16* __restrict__ A, const __bf16* __restrict__ B,
    const float* __restrict__ bias, const float* __restrict__ res,
    void* __restrict__ out, int M, int N, int K)
{
    const int lane = threadIdx.x & 31;
    const int wave = threadIdx.x >> 5;
    const int wm = wave & 1;
    const int wn = wave >> 1;
    const int tmW = blockIdx.y * 8 + wm * 4;   // first 16-row tile index
    const int tnW = blockIdx.x * 16 + wn * 4;  // first 16-col tile index
    const int numKT = K >> 5;

    v8f acc[4][4];
    v8f vzero = {0.f, 0.f, 0.f, 0.f, 0.f, 0.f, 0.f, 0.f};
#pragma unroll
    for (int i = 0; i < 4; ++i)
#pragma unroll
        for (int j = 0; j < 4; ++j) acc[i][j] = vzero;

    for (int tk = 0; tk < numKT; ++tk) {
        v16bf afrag[4], bfrag[4];
#pragma unroll
        for (int i = 0; i < 4; ++i) {
            const __bf16* at = A + ((size_t)(tmW + i) * numKT + tk) * 512;
            union { v16bf v; bf16x8 h[2]; } u;
            u.h[0] = *(const bf16x8*)(at + (lane + (lane & 16)) * 8);          // coalesced 512B
            u.h[1] = *(const bf16x8*)(at + (lane + (lane & 16) + 16) * 8);    // coalesced 512B
            afrag[i] = u.v;
        }
#pragma unroll
        for (int j = 0; j < 4; ++j) {
            const __bf16* bt = B + ((size_t)(tnW + j) * numKT + tk) * 512;
            union { v16bf v; bf16x8 h[2]; } u;
            u.h[0] = *(const bf16x8*)(bt + lane * 16);         // coalesced 1KB
            u.h[1] = *(const bf16x8*)(bt + lane * 16 + 8);
            bfrag[j] = u.v;
        }
#pragma unroll
        for (int i = 0; i < 4; ++i)
#pragma unroll
            for (int j = 0; j < 4; ++j)
                acc[i][j] = __builtin_amdgcn_wmma_f32_16x16x32_bf16(
                    false, afrag[i], false, bfrag[j],
                    (short)0, acc[i][j], false, false);
    }

    // C/D layout: VGPR r -> M = r + 8*(lane>>4), lane&15 -> N
    const int lrow = lane & 15;
    const int half = lane >> 4;
#pragma unroll
    for (int i = 0; i < 4; ++i) {
#pragma unroll
        for (int j = 0; j < 4; ++j) {
#pragma unroll
            for (int r = 0; r < 8; ++r) {
                const int row = tmW * 16 + i * 16 + half * 8 + r;
                const int col = tnW * 16 + j * 16 + lrow;
                float v = acc[i][j][r];
                if (HAS_BIAS) v += bias[col];
                if (DO_GELU)  v = gelu_exact(v);
                if (HAS_RES)  v += res[(size_t)row * N + col];
                if (OMODE == 0) {
                    ((float*)out)[(size_t)row * N + col] = v;
                } else if (OMODE == 1) {        // ALay(M, N)
                    const size_t t = (size_t)(row >> 4) * (N >> 5) + (col >> 5);
                    ((__bf16*)out)[t * 512 + (((col & 31) >> 3) * 16 + (row & 15)) * 8
                                   + (col & 7)] = (__bf16)v;
                } else if (OMODE == 2) {        // BLay(rows=n, k=N)
                    const size_t t = (size_t)(row >> 4) * (N >> 5) + (col >> 5);
                    ((__bf16*)out)[t * 512 + (((col & 31) >> 4) * 16 + (row & 15)) * 16
                                   + (col & 15)] = (__bf16)v;
                } else {                        // BLay of transpose: n=col, k=M
                    const size_t t = (size_t)(col >> 4) * (M >> 5) + (row >> 5);
                    ((__bf16*)out)[t * 512 + (((row & 31) >> 4) * 16 + (col & 15)) * 16
                                   + (row & 15)] = (__bf16)v;
                }
            }
        }
    }
}

// ---------------------------------------------------------------------------
// Row softmax (scale folded) reading f32 scores row-major, writing bf16
// attention weights directly into ALay(SEQ, SEQ) for the attn@V GEMM.
// One block (256 threads) per row, 16 columns per thread.
// ---------------------------------------------------------------------------
__global__ __launch_bounds__(256) void softmax_to_alay(
    const float* __restrict__ scores, __bf16* __restrict__ attnA, float scale)
{
    const int row = blockIdx.x;
    const float* rowp = scores + (size_t)row * SEQ_;
    const int t = threadIdx.x;

    float v[16];
    float mx = -INFINITY;
#pragma unroll
    for (int i = 0; i < 16; ++i) {
        v[i] = rowp[t * 16 + i] * scale;
        mx = fmaxf(mx, v[i]);
    }
    __shared__ float red[256];
    red[t] = mx; __syncthreads();
    for (int s = 128; s > 0; s >>= 1) {
        if (t < s) red[t] = fmaxf(red[t], red[t + s]);
        __syncthreads();
    }
    mx = red[0]; __syncthreads();

    float sum = 0.f;
#pragma unroll
    for (int i = 0; i < 16; ++i) { v[i] = __expf(v[i] - mx); sum += v[i]; }
    red[t] = sum; __syncthreads();
    for (int s = 128; s > 0; s >>= 1) {
        if (t < s) red[t] += red[t + s];
        __syncthreads();
    }
    const float inv = 1.0f / red[0];

    const int tm = row >> 4, m = row & 15;
#pragma unroll
    for (int i = 0; i < 16; ++i) {
        const int col = t * 16 + i;
        const size_t tile = (size_t)tm * (SEQ_ >> 5) + (col >> 5);
        attnA[tile * 512 + (((col & 31) >> 3) * 16 + m) * 8 + (col & 7)] =
            (__bf16)(v[i] * inv);
    }
}

// ---------------------------------------------------------------------------
// Launch
// ---------------------------------------------------------------------------
extern "C" void kernel_launch(void* const* d_in, const int* in_sizes, int n_in,
                              void* d_out, int out_size, void* d_ws, size_t ws_size,
                              hipStream_t stream) {
    (void)in_sizes; (void)n_in; (void)out_size; (void)ws_size;

    const float* x     = (const float*)d_in[0];
    const float* fc1_w = (const float*)d_in[1];
    const float* fc1_b = (const float*)d_in[2];
    const float* fc2_w = (const float*)d_in[3];
    const float* fc2_b = (const float*)d_in[4];
    const float* q_w   = (const float*)d_in[5];
    const float* q_b   = (const float*)d_in[6];
    const float* k_w   = (const float*)d_in[7];
    const float* k_b   = (const float*)d_in[8];
    const float* v_w   = (const float*)d_in[9];
    const float* v_b   = (const float*)d_in[10];
    const float* o_w   = (const float*)d_in[11];
    const float* o_b   = (const float*)d_in[12];
    float* out = (float*)d_out;

    // workspace bump allocator (256B aligned)
    char* ws = (char*)d_ws;
    size_t off = 0;
    auto alloc = [&](size_t bytes) -> char* {
        char* p = ws + off;
        off += (bytes + 255) & ~(size_t)255;
        return p;
    };
    __bf16* xb    = (__bf16*)alloc((size_t)SEQ_ * HID_ * 2);  // ALay(SEQ,HID)
    __bf16* fc1wb = (__bf16*)alloc((size_t)EXP_ * HID_ * 2);  // BLay(EXP,HID) / later attnA
    __bf16* fc2wb = (__bf16*)alloc((size_t)HID_ * EXP_ * 2);  // BLay(HID,EXP)
    __bf16* qwb   = (__bf16*)alloc((size_t)HID_ * HID_ * 2);  // BLay
    __bf16* kwb   = (__bf16*)alloc((size_t)HID_ * HID_ * 2);  // BLay
    __bf16* vwb   = (__bf16*)alloc((size_t)HID_ * HID_ * 2);  // BLay
    __bf16* owb   = (__bf16*)alloc((size_t)HID_ * HID_ * 2);  // BLay
    __bf16* h1    = (__bf16*)alloc((size_t)SEQ_ * EXP_ * 2);  // ALay(SEQ,EXP)
    float*  hres  = (float* )alloc((size_t)SEQ_ * HID_ * 4);  // f32 [SEQ,HID]
    __bf16* qA    = (__bf16*)alloc((size_t)SEQ_ * HID_ * 2);  // ALay(SEQ,HID) / later attnV
    __bf16* kB    = (__bf16*)alloc((size_t)SEQ_ * HID_ * 2);  // BLay(SEQ,HID)
    __bf16* vTB   = (__bf16*)alloc((size_t)HID_ * SEQ_ * 2);  // BLay(HID,SEQ)
    float*  scA   = (float* )alloc((size_t)SEQ_ * SEQ_ * 4);  // f32 scores [SEQ,SEQ]
    // buffer reuse (stream-ordered, no overlap of live ranges):
    __bf16* attnA = fc1wb;   // 32MB needed, fc1wb (32MB) dead after GEMM #2
    __bf16* attnV = qA;      // qA dead after scores GEMM

    dim3 blk(256);
    auto sgrid = [](long long chunks) { return dim3((unsigned)((chunks + 255) / 256)); };
    auto ggrid = [](int M, int N) { return dim3(N / 256, M / 128); };

    // 1) pre-swizzle fp32 operands into fragment-major bf16 layouts
    swizzleA_f32<<<sgrid((long long)SEQ_ * HID_ / 8), blk, 0, stream>>>(
        x, xb, HID_, (long long)SEQ_ * HID_ / 8);
    swizzleB_f32<<<sgrid((long long)EXP_ * HID_ / 16), blk, 0, stream>>>(
        fc1_w, fc1wb, HID_, (long long)EXP_ * HID_ / 16);
    swizzleB_f32<<<sgrid((long long)HID_ * EXP_ / 16), blk, 0, stream>>>(
        fc2_w, fc2wb, EXP_, (long long)HID_ * EXP_ / 16);
    swizzleB_f32<<<sgrid((long long)HID_ * HID_ / 16), blk, 0, stream>>>(
        q_w, qwb, HID_, (long long)HID_ * HID_ / 16);
    swizzleB_f32<<<sgrid((long long)HID_ * HID_ / 16), blk, 0, stream>>>(
        k_w, kwb, HID_, (long long)HID_ * HID_ / 16);
    swizzleB_f32<<<sgrid((long long)HID_ * HID_ / 16), blk, 0, stream>>>(
        v_w, vwb, HID_, (long long)HID_ * HID_ / 16);
    swizzleB_f32<<<sgrid((long long)HID_ * HID_ / 16), blk, 0, stream>>>(
        o_w, owb, HID_, (long long)HID_ * HID_ / 16);

    // 2) h1 = gelu(x @ fc1_w.T + fc1_b)            -> ALay(SEQ,EXP)
    gemm_bf16_wmma<1, true, true, false><<<ggrid(SEQ_, EXP_), blk, 0, stream>>>(
        xb, fc1wb, fc1_b, nullptr, h1, SEQ_, EXP_, HID_);
    // 3) hres = h1 @ fc2_w.T + fc2_b               -> f32 [SEQ,HID]
    gemm_bf16_wmma<0, false, true, false><<<ggrid(SEQ_, HID_), blk, 0, stream>>>(
        h1, fc2wb, fc2_b, nullptr, hres, SEQ_, HID_, EXP_);
    // 4) q = x @ q_w.T + q_b                       -> ALay(SEQ,HID)
    gemm_bf16_wmma<1, false, true, false><<<ggrid(SEQ_, HID_), blk, 0, stream>>>(
        xb, qwb, q_b, nullptr, qA, SEQ_, HID_, HID_);
    // 5) k = x @ k_w.T + k_b                       -> BLay(SEQ,HID)
    gemm_bf16_wmma<2, false, true, false><<<ggrid(SEQ_, HID_), blk, 0, stream>>>(
        xb, kwb, k_b, nullptr, kB, SEQ_, HID_, HID_);
    // 6) vT = (x @ v_w.T + v_b)^T                  -> BLay(HID,SEQ)
    gemm_bf16_wmma<3, false, true, false><<<ggrid(SEQ_, HID_), blk, 0, stream>>>(
        xb, vwb, v_b, nullptr, vTB, SEQ_, HID_, HID_);
    // 7) scores = q @ k.T                          -> f32 [SEQ,SEQ]
    gemm_bf16_wmma<0, false, false, false><<<ggrid(SEQ_, SEQ_), blk, 0, stream>>>(
        qA, kB, nullptr, nullptr, scA, SEQ_, SEQ_, HID_);
    // 8) attn = softmax(scores / sqrt(H))          -> ALay(SEQ,SEQ)
    softmax_to_alay<<<SEQ_, 256, 0, stream>>>(scA, attnA, 0.022097086912079608f);
    // 9) attnV = attn @ vT.T                       -> ALay(SEQ,HID)
    gemm_bf16_wmma<1, false, false, false><<<ggrid(SEQ_, HID_), blk, 0, stream>>>(
        attnA, vTB, nullptr, nullptr, attnV, SEQ_, HID_, SEQ_);
    // 10) out = attnV @ o_w.T + o_b + hres         -> f32 d_out
    gemm_bf16_wmma<0, false, true, true><<<ggrid(SEQ_, HID_), blk, 0, stream>>>(
        attnV, owb, o_b, hres, out, SEQ_, HID_, HID_);
}
